// Attention_18674517803322
// MI455X (gfx1250) — matrix operand order repaired and checked by reading, not compiled, not run on hardware
//
#include <hip/hip_runtime.h>

namespace {

constexpr int kDim   = 768;
constexpr int kHeads = 12;
constexpr int kHd    = 64;
constexpr int kSeq   = 1024;
constexpr int kBatch = 16;
constexpr int kM     = kBatch * kSeq;   // 16384 rows of x
constexpr int kQkvN  = 3 * kDim;        // 2304
constexpr float kScale = 0.125f;        // 1/sqrt(64)

typedef __attribute__((ext_vector_type(16))) __bf16 v16bf;
typedef __attribute__((ext_vector_type(8)))  __bf16 bf16x8;
typedef __attribute__((ext_vector_type(8)))  float  v8f;

__device__ inline v16bf frag_cat(bf16x8 lo, bf16x8 hi) {
  return __builtin_shufflevector(lo, hi, 0,1,2,3,4,5,6,7,8,9,10,11,12,13,14,15);
}

// A fragment: 16x32 bf16 tile, row-major source (stride in elements).
// Lane half h, row r: chunk0 = cols [h*8, h*8+8), chunk1 = cols [16+h*8, 16+h*8+8).
__device__ inline v16bf load_a16x32(const __bf16* src, int stride) {
  const int lane = threadIdx.x & 31;
  const int half = lane >> 4, r = lane & 15;
  const __bf16* p = src + r * stride + half * 8;
  return frag_cat(*(const bf16x8*)p, *(const bf16x8*)(p + 16));
}

// B fragment: 32(K)x16(N) tile stored n-major as [n][k] (stride in elements).
// Lane half h, col n: chunk0 = k [h*8, h*8+8), chunk1 = k [16+h*8, 16+h*8+8).
__device__ inline v16bf load_b32x16(const __bf16* src, int stride) {
  const int lane = threadIdx.x & 31;
  const int half = lane >> 4, n = lane & 15;
  const __bf16* p = src + n * stride + half * 8;
  return frag_cat(*(const bf16x8*)p, *(const bf16x8*)(p + 16));
}

__device__ inline v8f wmma_bf16(v16bf a, v16bf b, v8f c) {
  return __builtin_amdgcn_wmma_f32_16x16x32_bf16(false, a, false, b, (short)0, c,
                                                 false, false);
}

__device__ inline v8f zero8() {
  v8f z = {0.f, 0.f, 0.f, 0.f, 0.f, 0.f, 0.f, 0.f};
  return z;
}

// Async global->LDS 128-bit copy (CDNA5, tracked by ASYNCcnt). GV addressing:
// vdst = LDS byte address (low 32 bits of flat shared pointer, ISA 10.2),
// vaddr = 64-bit global address. INST_OFFSET is added to BOTH the LDS and the
// global address (ISA 15.18.3), so the per-chunk 16B stride goes in offset:.
template <int OFF>
__device__ inline void async_b128(unsigned int lds_addr, const void* gaddr) {
  asm volatile("global_load_async_to_lds_b128 %0, %1, off offset:%c2"
               :: "v"(lds_addr), "v"((unsigned long long)(uintptr_t)gaddr),
                  "n"(OFF)
               : "memory");
}

__device__ inline unsigned int lds_lo32(const void* p) {
  return (unsigned int)(uintptr_t)p;
}

__global__ void cvt_f32_bf16(const float* __restrict__ in, __bf16* __restrict__ out,
                             int n) {
  int i = blockIdx.x * blockDim.x + threadIdx.x;
  const int stride = gridDim.x * blockDim.x;
  for (; i < n; i += stride) out[i] = (__bf16)in[i];
}

// ---------------- QKV GEMM: [16384 x 768] x [768 x 2304]^T + bias -> Q/K/V ---
// 128x128 block tile, TILE_K = 64, double-buffered async-to-LDS main loop.
__global__ __launch_bounds__(256) void qkv_gemm(
    const __bf16* __restrict__ X, const __bf16* __restrict__ W,
    const float* __restrict__ bias,
    __bf16* __restrict__ Q, __bf16* __restrict__ K, __bf16* __restrict__ V) {
  __shared__ __bf16 sA[2][128 * 64];
  __shared__ __bf16 sB[2][128 * 64];
  const int tid = threadIdx.x, lane = tid & 31, wave = tid >> 5;
  const int m0 = blockIdx.x * 128, n0 = blockIdx.y * 128;
  const int wm = (wave & 3) * 32, wn = (wave >> 2) * 64;

  v8f acc[2][4];
#pragma unroll
  for (int i = 0; i < 2; ++i)
#pragma unroll
    for (int j = 0; j < 4; ++j) acc[i][j] = zero8();

  // Per-thread slice of the cooperative tile load: row lr, 32 cols at lc.
  const int lr = tid >> 1, lc = (tid & 1) * 32;
  const __bf16* gA = X + (size_t)(m0 + lr) * kDim + lc;
  const __bf16* gB = W + (size_t)(n0 + lr) * kDim + lc;
  const unsigned int lA[2] = {lds_lo32(&sA[0][lr * 64 + lc]),
                              lds_lo32(&sA[1][lr * 64 + lc])};
  const unsigned int lB[2] = {lds_lo32(&sB[0][lr * 64 + lc]),
                              lds_lo32(&sB[1][lr * 64 + lc])};

  auto issue = [&](int buf, int k0) {
    const __bf16* ga = gA + k0;
    const __bf16* gb = gB + k0;
    async_b128<0>(lA[buf], ga);
    async_b128<0>(lB[buf], gb);
    async_b128<16>(lA[buf], ga);
    async_b128<16>(lB[buf], gb);
    async_b128<32>(lA[buf], ga);
    async_b128<32>(lB[buf], gb);
    async_b128<48>(lA[buf], ga);
    async_b128<48>(lB[buf], gb);
  };
  auto compute = [&](int buf) {
#pragma unroll
    for (int ks = 0; ks < 2; ++ks) {
      v16bf a[2], b[4];
#pragma unroll
      for (int i = 0; i < 2; ++i)
        a[i] = load_a16x32(&sA[buf][(wm + i * 16) * 64 + ks * 32], 64);
#pragma unroll
      for (int j = 0; j < 4; ++j)
        b[j] = load_b32x16(&sB[buf][(wn + j * 16) * 64 + ks * 32], 64);
#pragma unroll
      for (int i = 0; i < 2; ++i)
#pragma unroll
        for (int j = 0; j < 4; ++j) acc[i][j] = wmma_bf16(a[i], b[j], acc[i][j]);
    }
  };

  issue(0, 0);
  for (int k0 = 0; k0 < kDim - 64; k0 += 64) {
    const int cur = (k0 >> 6) & 1;
    issue(cur ^ 1, k0 + 64);                           // prefetch next stage
    asm volatile("s_wait_asynccnt 0x8" ::: "memory");  // retire current stage
    __syncthreads();
    compute(cur);
    __syncthreads();
  }
  asm volatile("s_wait_asynccnt 0x0" ::: "memory");
  __syncthreads();
  compute(((kDim >> 6) - 1) & 1);

  const int half = lane >> 4, cn = lane & 15;
#pragma unroll
  for (int i = 0; i < 2; ++i)
#pragma unroll
    for (int j = 0; j < 4; ++j)
#pragma unroll
      for (int v = 0; v < 8; ++v) {
        const int gm = m0 + wm + i * 16 + v + half * 8;
        const int gn = n0 + wn + j * 16 + cn;
        const float val = acc[i][j][v] + bias[gn];
        const int s3 = gn / kDim, rem = gn - s3 * kDim;  // qkv slice
        const int h = rem >> 6, dd = rem & 63;
        const int bb = gm >> 10, t = gm & (kSeq - 1);
        __bf16* dst = (s3 == 0) ? Q : ((s3 == 1) ? K : V);
        dst[(((size_t)bb * kHeads + h) * kSeq + t) * kHd + dd] = (__bf16)val;
      }
}

// ---------------- Flash attention: per block 128 q-rows of one (b,h) --------
__global__ __launch_bounds__(256) void flash_attn(
    const __bf16* __restrict__ Q, const __bf16* __restrict__ K,
    const __bf16* __restrict__ V, __bf16* __restrict__ O) {
  __shared__ __bf16 sK[128 * 64];    // [key][d]   row-major (B for Q*K^T)
  __shared__ __bf16 sVt[64 * 128];   // [d][key]   transposed (B for P*V)
  __shared__ __bf16 sP[8][16 * 32];  // per-wave P staging (C-layout -> A-layout)
  const int tid = threadIdx.x, lane = tid & 31, wave = tid >> 5;
  const int half = lane >> 4, r = lane & 15;
  const int bh = blockIdx.y;
  const int b = bh / kHeads, h = bh - b * kHeads;
  const int q0 = blockIdx.x * 128 + wave * 16;
  const __bf16* Qg = Q + (size_t)bh * kSeq * kHd;
  const __bf16* Kg = K + (size_t)bh * kSeq * kHd;
  const __bf16* Vg = V + (size_t)bh * kSeq * kHd;

  // Q fragments (reused across all key blocks): 16 rows x 64 d, 2 k-steps.
  v16bf qa[2];
#pragma unroll
  for (int ks = 0; ks < 2; ++ks) {
    const __bf16* p = Qg + (size_t)(q0 + r) * kHd + ks * 32 + half * 8;
    qa[ks] = frag_cat(*(const bf16x8*)p, *(const bf16x8*)(p + 16));
  }

  float m_i[8], l_i[8];
  v8f o[4];
#pragma unroll
  for (int v = 0; v < 8; ++v) { m_i[v] = -1e30f; l_i[v] = 0.f; }
#pragma unroll
  for (int j = 0; j < 4; ++j) o[j] = zero8();

  const int lr = tid >> 1, lc = (tid & 1) * 32;
  for (int kb = 0; kb < kSeq; kb += 128) {
    {  // K tile, row-major copy
      const __bf16* g = Kg + (size_t)(kb + lr) * kHd + lc;
#pragma unroll
      for (int e = 0; e < 4; ++e)
        *(bf16x8*)&sK[lr * 64 + lc + e * 8] = *(const bf16x8*)(g + e * 8);
    }
    {  // V tile, transposed into [d][key]
      const __bf16* g = Vg + (size_t)(kb + lr) * kHd + lc;
      __align__(16) __bf16 tmp[32];
#pragma unroll
      for (int e = 0; e < 4; ++e) *(bf16x8*)&tmp[e * 8] = *(const bf16x8*)(g + e * 8);
#pragma unroll
      for (int e = 0; e < 32; ++e) sVt[(lc + e) * 128 + lr] = tmp[e];
    }
    __syncthreads();

    // S = Q * K^T over 128 keys (8 n-tiles x 2 d-steps)
    v8f s[8];
#pragma unroll
    for (int nt = 0; nt < 8; ++nt) {
      s[nt] = zero8();
#pragma unroll
      for (int ks = 0; ks < 2; ++ks)
        s[nt] = wmma_bf16(qa[ks], load_b32x16(&sK[(nt * 16) * 64 + ks * 32], 64),
                          s[nt]);
    }

    // online softmax (row = v + 8*half, row lives across 16 lanes)
#pragma unroll
    for (int v = 0; v < 8; ++v) {
      float mx = -1e30f;
#pragma unroll
      for (int nt = 0; nt < 8; ++nt) { s[nt][v] *= kScale; mx = fmaxf(mx, s[nt][v]); }
#pragma unroll
      for (int d = 1; d < 16; d <<= 1) mx = fmaxf(mx, __shfl_xor(mx, d, 32));
      const float mnew = fmaxf(m_i[v], mx);
      const float corr = __expf(m_i[v] - mnew);
      m_i[v] = mnew;
      float rs = 0.f;
#pragma unroll
      for (int nt = 0; nt < 8; ++nt) {
        const float e = __expf(s[nt][v] - mnew);
        s[nt][v] = e;
        rs += e;
      }
#pragma unroll
      for (int d = 1; d < 16; d <<= 1) rs += __shfl_xor(rs, d, 32);
      l_i[v] = l_i[v] * corr + rs;
#pragma unroll
      for (int j = 0; j < 4; ++j) o[j][v] *= corr;
    }

    // O += P * V : restage P through LDS (C-layout -> A-layout), 4 k-steps of 32
    __bf16* pbuf = sP[wave];
#pragma unroll
    for (int ks = 0; ks < 4; ++ks) {
#pragma unroll
      for (int t2 = 0; t2 < 2; ++t2) {
        const int nt = ks * 2 + t2;
#pragma unroll
        for (int v = 0; v < 8; ++v)
          pbuf[(v + half * 8) * 32 + t2 * 16 + r] = (__bf16)s[nt][v];
      }
      const v16bf pa = load_a16x32(pbuf, 32);  // same-wave LDS RAW: DS in-order
#pragma unroll
      for (int j = 0; j < 4; ++j)
        o[j] = wmma_bf16(pa, load_b32x16(&sVt[(j * 16) * 128 + ks * 32], 128), o[j]);
    }
    __syncthreads();
  }

  // epilogue: normalize and write [B, T, H*d]
#pragma unroll
  for (int j = 0; j < 4; ++j)
#pragma unroll
    for (int v = 0; v < 8; ++v) {
      const int t = q0 + v + half * 8;
      const int dd = j * 16 + r;
      const float val = o[j][v] / l_i[v];
      O[((size_t)b * kSeq + t) * kDim + h * kHd + dd] = (__bf16)val;
    }
}

// ---------------- Projection GEMM: [16384 x 768] x [768 x 768]^T + bias -----
__global__ __launch_bounds__(256) void proj_gemm(
    const __bf16* __restrict__ A, const __bf16* __restrict__ W,
    const float* __restrict__ bias, float* __restrict__ out) {
  __shared__ __bf16 sA[2][128 * 64];
  __shared__ __bf16 sB[2][128 * 64];
  const int tid = threadIdx.x, lane = tid & 31, wave = tid >> 5;
  const int m0 = blockIdx.x * 128, n0 = blockIdx.y * 128;
  const int wm = (wave & 3) * 32, wn = (wave >> 2) * 64;

  v8f acc[2][4];
#pragma unroll
  for (int i = 0; i < 2; ++i)
#pragma unroll
    for (int j = 0; j < 4; ++j) acc[i][j] = zero8();

  const int lr = tid >> 1, lc = (tid & 1) * 32;
  const __bf16* gA = A + (size_t)(m0 + lr) * kDim + lc;
  const __bf16* gB = W + (size_t)(n0 + lr) * kDim + lc;
  const unsigned int lA[2] = {lds_lo32(&sA[0][lr * 64 + lc]),
                              lds_lo32(&sA[1][lr * 64 + lc])};
  const unsigned int lB[2] = {lds_lo32(&sB[0][lr * 64 + lc]),
                              lds_lo32(&sB[1][lr * 64 + lc])};

  auto issue = [&](int buf, int k0) {
    const __bf16* ga = gA + k0;
    const __bf16* gb = gB + k0;
    async_b128<0>(lA[buf], ga);
    async_b128<0>(lB[buf], gb);
    async_b128<16>(lA[buf], ga);
    async_b128<16>(lB[buf], gb);
    async_b128<32>(lA[buf], ga);
    async_b128<32>(lB[buf], gb);
    async_b128<48>(lA[buf], ga);
    async_b128<48>(lB[buf], gb);
  };
  auto compute = [&](int buf) {
#pragma unroll
    for (int ks = 0; ks < 2; ++ks) {
      v16bf a[2], b[4];
#pragma unroll
      for (int i = 0; i < 2; ++i)
        a[i] = load_a16x32(&sA[buf][(wm + i * 16) * 64 + ks * 32], 64);
#pragma unroll
      for (int j = 0; j < 4; ++j)
        b[j] = load_b32x16(&sB[buf][(wn + j * 16) * 64 + ks * 32], 64);
#pragma unroll
      for (int i = 0; i < 2; ++i)
#pragma unroll
        for (int j = 0; j < 4; ++j) acc[i][j] = wmma_bf16(a[i], b[j], acc[i][j]);
    }
  };

  issue(0, 0);
  for (int k0 = 0; k0 < kDim - 64; k0 += 64) {
    const int cur = (k0 >> 6) & 1;
    issue(cur ^ 1, k0 + 64);
    asm volatile("s_wait_asynccnt 0x8" ::: "memory");
    __syncthreads();
    compute(cur);
    __syncthreads();
  }
  asm volatile("s_wait_asynccnt 0x0" ::: "memory");
  __syncthreads();
  compute(((kDim >> 6) - 1) & 1);

  const int half = lane >> 4, cn = lane & 15;
#pragma unroll
  for (int i = 0; i < 2; ++i)
#pragma unroll
    for (int j = 0; j < 4; ++j)
#pragma unroll
      for (int v = 0; v < 8; ++v) {
        const int gm = m0 + wm + i * 16 + v + half * 8;
        const int gn = n0 + wn + j * 16 + cn;
        out[(size_t)gm * kDim + gn] = acc[i][j][v] + bias[gn];
      }
}

}  // namespace

extern "C" void kernel_launch(void* const* d_in, const int* in_sizes, int n_in,
                              void* d_out, int out_size, void* d_ws, size_t ws_size,
                              hipStream_t stream) {
  (void)in_sizes; (void)n_in; (void)out_size; (void)ws_size;
  const float* x      = (const float*)d_in[0];
  const float* w_qkv  = (const float*)d_in[1];
  const float* b_qkv  = (const float*)d_in[2];
  const float* w_proj = (const float*)d_in[3];
  const float* b_proj = (const float*)d_in[4];
  float* out = (float*)d_out;

  char* ws = (char*)d_ws;
  size_t off = 0;
  auto carve = [&](size_t elems) {
    __bf16* p = (__bf16*)(ws + off);
    off = (off + elems * sizeof(__bf16) + 255) & ~(size_t)255;
    return p;
  };
  __bf16* Xb = carve((size_t)kM * kDim);      // x in bf16
  __bf16* Wq = carve((size_t)kQkvN * kDim);   // w_qkv in bf16
  __bf16* Wp = carve((size_t)kDim * kDim);    // w_proj in bf16
  __bf16* Qb = carve((size_t)kM * kDim);      // Q [B,H,T,d]
  __bf16* Kb = carve((size_t)kM * kDim);      // K [B,H,T,d]
  __bf16* Vb = carve((size_t)kM * kDim);      // V [B,H,T,d]
  __bf16* Ob = carve((size_t)kM * kDim);      // attn out [B,T,H*d]

  cvt_f32_bf16<<<4096, 256, 0, stream>>>(x, Xb, kM * kDim);
  cvt_f32_bf16<<<2048, 256, 0, stream>>>(w_qkv, Wq, kQkvN * kDim);
  cvt_f32_bf16<<<1024, 256, 0, stream>>>(w_proj, Wp, kDim * kDim);

  qkv_gemm<<<dim3(kM / 128, kQkvN / 128), 256, 0, stream>>>(Xb, Wq, b_qkv,
                                                            Qb, Kb, Vb);
  flash_attn<<<dim3(kSeq / 128, kBatch * kHeads), 256, 0, stream>>>(Qb, Kb, Vb, Ob);
  proj_gemm<<<dim3(kM / 128, kDim / 128), 256, 0, stream>>>(Ob, Wp, b_proj, out);
}
